// BEVSDTransformerDecoder_48584670052381
// MI455X (gfx1250) — compile-verified, matrix operand order checked by reading, not run on hardware
//
#include <hip/hip_runtime.h>
#include <hip/hip_bf16.h>
#include <math.h>

// ---------------- constants (match reference hyper-params) ----------------
#define NHEAD 4
#define NLVL  4
#define NPIL  4
#define NPT   2
#define PPTS  (NPIL * NPT)      // 8
#define NQ    1024
#define NC    256
#define NCAM  6

__device__ __constant__ int c_LH[4] = {32, 16, 8, 4};
__device__ __constant__ int c_LW[4] = {88, 44, 22, 11};
// float offsets of each level inside the transposed feature buffer (6*H*W*256)
__device__ __constant__ int c_LB[4] = {0, 4325376, 5406720, 5677056};

typedef _Float16 f16_t;
typedef __attribute__((ext_vector_type(8)))  _Float16 v8h;
typedef __attribute__((ext_vector_type(16))) _Float16 v16h;
typedef __attribute__((ext_vector_type(8)))  float    v8f;

// ---------------- small utility kernels ----------------
__global__ void qp_kernel(const float* __restrict__ q,
                          const float* __restrict__ qp,
                          f16_t* __restrict__ out, int n) {
  int i = blockIdx.x * blockDim.x + threadIdx.x;
  if (i < n) out[i] = (f16_t)(q[i] + qp[i]);
}

// Pack a KxN f32 weight matrix into WMMA B-fragment order (f16):
// Bp[((nt*(K/32)+kt)*32 + lane)*16 + i]  so each lane loads one contiguous
// 32-byte fragment (two global_load_b128) per K-step.
__global__ void pack_b_kernel(const float* __restrict__ B, f16_t* __restrict__ Bp,
                              int K, int N) {
  int idx = blockIdx.x * blockDim.x + threadIdx.x;
  if (idx >= K * N) return;
  int i  = idx & 15;            // fragment element
  int l  = (idx >> 4) & 31;     // lane
  int kt = (idx >> 9) % (K >> 5);
  int nt = idx / ((K >> 5) << 9);
  int n  = (nt << 4) + (l & 15);
  int hi = (l >> 4) << 3;                          // 0 or 8
  int k  = (kt << 5) + hi + (i & 7) + ((i >> 3) << 4);  // ISA 16-bit 32x16 layout
  Bp[idx] = (f16_t)B[(size_t)k * N + n];
}

// ---------------- projection: ref points -> per-cam normalized coords ------
__global__ void proj_kernel(const float* __restrict__ rp,   // (1,4,32,32,3)
                            const float* __restrict__ l2i,  // (1,6,4,4)
                            float* __restrict__ gxy,        // (6,1024,4,2)
                            float* __restrict__ qmask) {    // (6,1024)
  int q = blockIdx.x * blockDim.x + threadIdx.x;
  if (q >= NQ) return;
  int hb = q >> 5, wb = q & 31;
  float X[NPIL], Y[NPIL], Z[NPIL];
  for (int p = 0; p < NPIL; ++p) {
    const float* r = rp + ((p * 32 + hb) * 32 + wb) * 3;
    X[p] = r[0] * 102.4f - 51.2f;
    Y[p] = r[1] * 102.4f - 51.2f;
    Z[p] = r[2] * 8.0f   - 5.0f;
  }
  for (int cam = 0; cam < NCAM; ++cam) {
    const float* M = l2i + cam * 16;
    int any = 0;
    for (int p = 0; p < NPIL; ++p) {
      float u = M[0] * X[p] + M[1] * Y[p] + M[2]  * Z[p] + M[3];
      float v = M[4] * X[p] + M[5] * Y[p] + M[6]  * Z[p] + M[7];
      float d = M[8] * X[p] + M[9] * Y[p] + M[10] * Z[p] + M[11];
      float dn = fmaxf(d, 1e-5f);
      float gx = (u / dn) * (2.0f / 704.0f) - 1.0f;
      float gy = (v / dn) * (2.0f / 256.0f) - 1.0f;
      int idx = (((cam << 10) + q) * NPIL + p) * 2;
      gxy[idx + 0] = gx;
      gxy[idx + 1] = gy;
      any |= (d > 1e-5f) && (gx > -1.f) && (gx < 1.f) && (gy > -1.f) && (gy < 1.f);
    }
    qmask[(cam << 10) + q] = any ? 1.0f : 0.0f;
  }
}

// ---------------- feature transpose (cam,C,H,W) -> (cam,H,W,C), LDS-tiled ---
__global__ __launch_bounds__(256)
void transpose_feat_kernel(const float* __restrict__ src, float* __restrict__ dst,
                           int H, int W, int nxt) {
  __shared__ float tile[32 * 257];
  int bid = blockIdx.x;
  int xt   = bid % nxt;
  int rest = bid / nxt;
  int y    = rest % H;
  int cam  = rest / H;
  int tid  = threadIdx.x;
  int xl   = tid & 31;          // x within tile
  int cgrp = tid >> 5;          // 0..7, channel group
  int x = xt * 32 + xl;
  const float* sp = src + ((size_t)(cam * NC) * H + y) * W;
  if (x < W) {
    #pragma unroll 4
    for (int cc = 0; cc < 32; ++cc) {
      int c = cc * 8 + cgrp;
      tile[xl * 257 + c] = sp[(size_t)c * H * W + x];  // coalesced over x
    }
  }
  __syncthreads();
  int c = tid;
  float* dp = dst + ((size_t)(cam * H + y) * W) * NC;
  #pragma unroll 4
  for (int xx = 0; xx < 32; ++xx) {
    int xg = xt * 32 + xx;
    if (xg < W) dp[(size_t)xg * NC + c] = tile[xx * 257 + c];  // coalesced over c
  }
}

// ---------------- WMMA f16 GEMM: D = A(MxK) @ B(KxN) + bias [+sigmoid/resid]
// MODE 0: +bias ; MODE 1: sigmoid(+bias) ; MODE 2: +bias +resid
// Bp is pre-packed into fragment order by pack_b_kernel. K unrolled at
// compile time -> 8 back-to-back v_wmma per wave. blockDim=(32,4): each wave
// owns one 16-wide N tile (wave-uniform guard keeps EXEC all-ones).
template <int K, int MODE>
__global__ __launch_bounds__(128)
void wmma_gemm_kernel(const f16_t* __restrict__ A, const v16h* __restrict__ Bp,
                      const float* __restrict__ bias, const float* __restrict__ resid,
                      float* __restrict__ D, int N) {
  constexpr int KT = K / 32;
  int nt = blockIdx.x * blockDim.y + threadIdx.y;   // N tile (wave-uniform)
  if ((nt << 4) >= N) return;
  int row0 = blockIdx.y << 4;
  int lane = threadIdx.x & 31;
  int t  = lane & 15;          // A: M index / D: N index
  int hi = (lane >> 4) << 3;   // 0 or 8
  const f16_t* Ap = A + (size_t)(row0 + t) * K + hi;
  const v16h*  Bw = Bp + (size_t)nt * KT * 32 + lane;
  v8f c = {};
  #pragma unroll
  for (int kt = 0; kt < KT; ++kt) {
    v8h alo = *(const v8h*)(Ap + (kt << 5));        // 16B contiguous
    v8h ahi = *(const v8h*)(Ap + (kt << 5) + 16);   // 16B contiguous
    v16h a;
    #pragma unroll
    for (int i = 0; i < 8; ++i) { a[i] = alo[i]; a[8 + i] = ahi[i]; }
    v16h b = Bw[(size_t)kt << 5];                   // 32B contiguous fragment
    c = __builtin_amdgcn_wmma_f32_16x16x32_f16(false, a, false, b,
                                               (short)0, c, false, false);
  }
  int n = (nt << 4) + t;
  float bn = bias[n];
  #pragma unroll
  for (int r = 0; r < 8; ++r) {        // C/D: VGPR r -> M, lane -> N
    int m = row0 + hi + r;
    float v = c[r] + bn;
    if (MODE == 1)      v = 1.0f / (1.0f + expf(-v));
    else if (MODE == 2) v += resid[(size_t)m * N + n];
    D[(size_t)m * N + n] = v;
  }
}

// ---------------- tap-list deformable sampling + attention reduce -----------
#define MAX_TAPS 3072   // 6 cam * 4 head * 8 pts * 4 lvl * 4 taps (upper bound)
__global__ __launch_bounds__(256)
void sample_reduce_kernel(const float* __restrict__ off,    // (1024,256)
                          const float* __restrict__ attn,   // (1024,128)
                          const float* __restrict__ gxy,    // (6,1024,4,2)
                          const float* __restrict__ qmask,  // (6,1024)
                          const float* __restrict__ featT,  // transposed pyramid
                          float* __restrict__ tmp,          // (1024,256)
                          f16_t* __restrict__ tmph) {
  __shared__ int   s_cnt;
  __shared__ int   s_tap[MAX_TAPS];
  __shared__ float s_w[MAX_TAPS];
  int q = blockIdx.x;
  int tid = threadIdx.x;
  if (tid == 0) s_cnt = 0;
  __syncthreads();

  // Phase A: 768 (cam,head,pt,lvl) combos, 3 per thread -> compacted tap list
  for (int i = 0; i < 3; ++i) {
    int combo = tid + (i << 8);
    if (combo >= NCAM * NHEAD * PPTS * NLVL) break;
    int l   = combo & 3;
    int p   = (combo >> 2) & 7;
    int h   = (combo >> 5) & 3;
    int cam = combo >> 7;
    if (qmask[(cam << 10) + q] == 0.0f) continue;   // masked camera: skip
    int pil = p >> 1, pt = p & 1;
    const float* g = gxy + (((cam << 10) + q) * NPIL + pil) * 2;
    int oi = (q << 8) + (((((h << 2) + l) << 2) + pil) * 2 + pt) * 2;
    float cx = g[0] + off[oi];
    float cy = g[1] + off[oi + 1];
    float aw = attn[(q << 7) + (h << 5) + (p << 2) + l];
    // clamp keeps int math safe; clamped values remain strictly out of image
    cx = fminf(fmaxf(cx, -8.0f), 8.0f);
    cy = fminf(fmaxf(cy, -8.0f), 8.0f);
    int Wl = c_LW[l], Hl = c_LH[l];
    float xf = (cx + 1.0f) * (Wl * 0.5f) - 0.5f;
    float yf = (cy + 1.0f) * (Hl * 0.5f) - 0.5f;
    float x0f = floorf(xf), y0f = floorf(yf);
    int x0 = (int)x0f, y0 = (int)y0f;
    float wx1 = xf - x0f, wy1 = yf - y0f;
    float wx0 = 1.0f - wx1, wy0 = 1.0f - wy1;
    float wt[4] = {wx0 * wy0, wx1 * wy0, wx0 * wy1, wx1 * wy1};
    int base = c_LB[l] + cam * Hl * Wl * NC;
    #pragma unroll
    for (int tp = 0; tp < 4; ++tp) {
      int ix = x0 + (tp & 1);
      int iy = y0 + (tp >> 1);
      float w = wt[tp] * aw;
      if (ix < 0 || ix >= Wl || iy < 0 || iy >= Hl || w == 0.0f) continue;
      int slot = atomicAdd(&s_cnt, 1);
      s_tap[slot] = base + ((iy * Wl + ix) << 8);
      s_w[slot]   = w;
    }
  }
  __syncthreads();

  // Phase B: every channel-thread streams the tap list (1KB coalesced / tap)
  int n = s_cnt;
  float acc = 0.0f;
  int t = 0;
  for (; t + 4 <= n; t += 4) {
    acc += s_w[t + 0] * featT[s_tap[t + 0] + tid];
    acc += s_w[t + 1] * featT[s_tap[t + 1] + tid];
    acc += s_w[t + 2] * featT[s_tap[t + 2] + tid];
    acc += s_w[t + 3] * featT[s_tap[t + 3] + tid];
  }
  for (; t < n; ++t) acc += s_w[t] * featT[s_tap[t] + tid];
  tmp[(q << 8) + tid]  = acc;
  tmph[(q << 8) + tid] = (f16_t)acc;
}

// ---------------- launcher ----------------
extern "C" void kernel_launch(void* const* d_in, const int* in_sizes, int n_in,
                              void* d_out, int out_size, void* d_ws, size_t ws_size,
                              hipStream_t stream) {
  (void)in_sizes; (void)n_in; (void)out_size; (void)ws_size;
  const float* query  = (const float*)d_in[0];
  const float* qpos   = (const float*)d_in[1];
  const float* rp     = (const float*)d_in[2];
  const float* l2i    = (const float*)d_in[3];
  const float* feat[4] = {(const float*)d_in[4], (const float*)d_in[5],
                          (const float*)d_in[6], (const float*)d_in[7]};
  const float* W_off  = (const float*)d_in[8];
  const float* b_off  = (const float*)d_in[9];
  const float* W_attn = (const float*)d_in[10];
  const float* b_attn = (const float*)d_in[11];
  const float* W_out  = (const float*)d_in[12];
  const float* b_out  = (const float*)d_in[13];

  // ---- workspace layout (bytes, all 32B-aligned) ----
  char* ws = (char*)d_ws;
  f16_t* qp_h    = (f16_t*)(ws + 0);              // 1024*256 half
  f16_t* Woff_p  = (f16_t*)(ws + 524288);         // 256*256 half (packed)
  f16_t* Wattn_p = (f16_t*)(ws + 655360);         // 256*128 half (packed)
  f16_t* Wout_p  = (f16_t*)(ws + 720896);         // 256*256 half (packed)
  float* off_f   = (float*)(ws + 851968);         // 1024*256 f32
  float* attn_f  = (float*)(ws + 1900544);        // 1024*128 f32
  float* gxy     = (float*)(ws + 2424832);        // 6*1024*4*2 f32
  float* qmask   = (float*)(ws + 2621440);        // 6*1024 f32
  float* tmp_f   = (float*)(ws + 2646016);        // 1024*256 f32
  f16_t* tmp_h   = (f16_t*)(ws + 3694592);        // 1024*256 half
  float* featT   = (float*)(ws + 4218880);        // 5744640 f32 (~22.9 MB)

  const int LH[4] = {32, 16, 8, 4};
  const int LW[4] = {88, 44, 22, 11};
  const int LB[4] = {0, 4325376, 5406720, 5677056};

  // weight packing (f32 -> f16 fragment layout) + qp
  pack_b_kernel<<<(65536 + 255) / 256, 256, 0, stream>>>(W_off,  Woff_p,  256, 256);
  pack_b_kernel<<<(32768 + 255) / 256, 256, 0, stream>>>(W_attn, Wattn_p, 256, 128);
  pack_b_kernel<<<(65536 + 255) / 256, 256, 0, stream>>>(W_out,  Wout_p,  256, 256);
  qp_kernel<<<NQ, NC, 0, stream>>>(query, qpos, qp_h, NQ * NC);

  // projection
  proj_kernel<<<NQ / 256, 256, 0, stream>>>(rp, l2i, gxy, qmask);

  // feature transpose per level
  for (int l = 0; l < 4; ++l) {
    int nxt = (LW[l] + 31) / 32;
    int blocks = NCAM * LH[l] * nxt;
    transpose_feat_kernel<<<blocks, 256, 0, stream>>>(feat[l], featT + LB[l],
                                                      LH[l], LW[l], nxt);
  }

  // off = qp @ W_off + b_off          (1024x256x256)
  wmma_gemm_kernel<256, 0><<<dim3(4, NQ / 16), dim3(32, 4), 0, stream>>>(
      qp_h, (const v16h*)Woff_p, b_off, nullptr, off_f, NC);
  // attnw = sigmoid(qp @ W_attn + b)  (1024x128x256)
  wmma_gemm_kernel<256, 1><<<dim3(2, NQ / 16), dim3(32, 4), 0, stream>>>(
      qp_h, (const v16h*)Wattn_p, b_attn, nullptr, attn_f, 128);

  // deformable sampling + attention-weighted reduce
  sample_reduce_kernel<<<NQ, 256, 0, stream>>>(off_f, attn_f, gxy, qmask,
                                               featT, tmp_f, tmp_h);

  // out = tmp @ W_out + b_out + query (1024x256x256)
  wmma_gemm_kernel<256, 2><<<dim3(4, NQ / 16), dim3(32, 4), 0, stream>>>(
      tmp_h, (const v16h*)Wout_p, b_out, query, (float*)d_out, NC);
}
